// LabelAwareClassifier_43361989820524
// MI455X (gfx1250) — matrix operand order repaired and checked by reading, not compiled, not run on hardware
//
#include <hip/hip_runtime.h>

// ---------------------------------------------------------------------------
// LabelAwareClassifier fused pipeline for gfx1250 (MI455X, wave32, WMMA)
//   B=16384, D=4096, E=C=128, H=8, DH=16
// Pipeline (all bf16 WMMA, f32 accumulate):
//   prep:  Wf/Wq/Wo/Wc -> bf16 ; k,v = label_emb@W{k,v}.T + b  -> Karr/Varr
//   fused: f = x@Wf^T+bf ; q=(f@Wq^T+bq)/4 ; per-head scores via padded-K
//          WMMA ; softmax (shfl) ; o = attn@V ; o2=o@Wo^T+bo ; out=o2@Wc^T+bc
// ---------------------------------------------------------------------------

typedef __attribute__((ext_vector_type(16))) __bf16 v16bf;
typedef __attribute__((ext_vector_type(8)))  __bf16 bf16x8;
typedef __attribute__((ext_vector_type(8)))  float  v8f;

#define B_ROWS 16384
#define D_DIM  4096
#define E_DIM  128
#define C_DIM  128

__device__ __forceinline__ __bf16 f2bf(float f) {
  unsigned u = __builtin_bit_cast(unsigned, f);
  unsigned r = (u + 0x7FFFu + ((u >> 16) & 1u)) >> 16;
  unsigned short s = (unsigned short)r;
  return __builtin_bit_cast(__bf16, s);
}

__device__ __forceinline__ v16bf pack16(bf16x8 lo, bf16x8 hi) {
  v16bf r;
#pragma unroll
  for (int i = 0; i < 8; ++i) { r[i] = lo[i]; r[i + 8] = hi[i]; }
  return r;
}

// B-fragment: lane holds 16 *consecutive* K values (K = laneHi*16 + e)
__device__ __forceinline__ v16bf loadB16(const __bf16* p) {
  const bf16x8* q = reinterpret_cast<const bf16x8*>(p);
  return pack16(q[0], q[1]);
}

// A-fragment from LDS row: K = laneHi*16 + e
__device__ __forceinline__ v16bf loadA_lds(const __bf16* row, int k0, int laneHi) {
  const bf16x8* lo = reinterpret_cast<const bf16x8*>(row + k0 + laneHi * 16);
  const bf16x8* hi = reinterpret_cast<const bf16x8*>(row + k0 + laneHi * 16 + 8);
  return pack16(lo[0], hi[0]);
}

__device__ __forceinline__ void put4(v16bf& a, int base, float4 f) {
  a[base + 0] = f2bf(f.x); a[base + 1] = f2bf(f.y);
  a[base + 2] = f2bf(f.z); a[base + 3] = f2bf(f.w);
}

__device__ __forceinline__ v8f wmma_bf16(v16bf a, v16bf b, v8f c) {
  return __builtin_amdgcn_wmma_f32_16x16x32_bf16(false, a, false, b,
                                                 (short)0, c, false, false);
}

// ---------------------------------------------------------------------------
// Prep kernels
// ---------------------------------------------------------------------------
__global__ void cvt_bf16_kernel(const float* __restrict__ src,
                                __bf16* __restrict__ dst, int n) {
  int i = blockIdx.x * blockDim.x + threadIdx.x;
  if (i < n) dst[i] = f2bf(src[i]);
}

// k = label_emb@Wk^T+bk, v = label_emb@Wv^T+bv, repacked for WMMA:
//  Karr[h][c][32]  bf16, head data at K-offset (h&1)*16, other half zero
//  Varr[h][d][128] bf16 (transposed V per head)
__global__ void prep_kv_kernel(const float* __restrict__ label_emb,
                               const float* __restrict__ Wk, const float* __restrict__ bk,
                               const float* __restrict__ Wv, const float* __restrict__ bv,
                               __bf16* __restrict__ Karr, __bf16* __restrict__ Varr) {
  int gid = blockIdx.x * blockDim.x + threadIdx.x;   // 0..16383
  int c = gid >> 7, e = gid & 127;
  const float* le = label_emb + (size_t)c * E_DIM;
  const float* wk = Wk + (size_t)e * E_DIM;
  const float* wv = Wv + (size_t)e * E_DIM;
  float ka = bk[e], va = bv[e];
#pragma unroll 8
  for (int i = 0; i < E_DIM; ++i) { float l = le[i]; ka += l * wk[i]; va += l * wv[i]; }
  int h = e >> 4, d = e & 15, off = (h & 1) * 16;
  Karr[((size_t)(h * 128 + c)) * 32 + off + d]        = f2bf(ka);
  Karr[((size_t)(h * 128 + c)) * 32 + (off ^ 16) + d] = f2bf(0.f);
  Varr[((size_t)(h * 16 + d)) * 128 + c]              = f2bf(va);
}

// ---------------------------------------------------------------------------
// Fused main kernel: 256 blocks x 128 threads (4 waves); wave = 16 rows.
// No barriers needed: each wave owns its 16 rows end-to-end (all reductions
// are along columns which the wave fully owns).
// ---------------------------------------------------------------------------
__global__ __launch_bounds__(128) void fused_kernel(
    const float* __restrict__ x,
    const float* __restrict__ bfp, const float* __restrict__ bqp,
    const float* __restrict__ bop, const float* __restrict__ bcp,
    const __bf16* __restrict__ Wfb, const __bf16* __restrict__ Wqb,
    const __bf16* __restrict__ Wob, const __bf16* __restrict__ Wcb,
    const __bf16* __restrict__ Karr, const __bf16* __restrict__ Varr,
    float* __restrict__ out) {
  const int lane   = threadIdx.x & 31;
  const int wave   = threadIdx.x >> 5;
  const int laneHi = lane >> 4;   // 0 or 1
  const int ln     = lane & 15;
  const int rowBase = blockIdx.x * 64 + wave * 16;

  __shared__ __align__(16) __bf16 bufA[4][16][128];
  __shared__ __align__(16) __bf16 bufB[4][16][128];
  __bf16 (*mybufA)[128] = bufA[wave];
  __bf16 (*mybufB)[128] = bufB[wave];

  float bfv[8], bqv[8], bov[8], bcv[8];
#pragma unroll
  for (int t = 0; t < 8; ++t) {
    bfv[t] = bfp[t * 16 + ln]; bqv[t] = bqp[t * 16 + ln];
    bov[t] = bop[t * 16 + ln]; bcv[t] = bcp[t * 16 + ln];
  }

  // ---------------- Stage 1: f = x @ Wf^T + bf  (K = 4096) ----------------
  v8f acc[8];
#pragma unroll
  for (int t = 0; t < 8; ++t)
#pragma unroll
    for (int r = 0; r < 8; ++r) acc[t][r] = bfv[t];

  const float* xrow = x + (size_t)(rowBase + ln) * D_DIM;
  for (int k0 = 0; k0 < D_DIM; k0 += 32) {
    int ka = k0 + laneHi * 16;
    const float4* p0 = reinterpret_cast<const float4*>(xrow + ka);
    const float4* p1 = reinterpret_cast<const float4*>(xrow + ka + 8);
    float4 a0 = p0[0], a1 = p0[1], a2 = p1[0], a3 = p1[1];
    v16bf A;
    put4(A, 0, a0); put4(A, 4, a1); put4(A, 8, a2); put4(A, 12, a3);
#pragma unroll
    for (int t = 0; t < 8; ++t) {
      const __bf16* wr = Wfb + (size_t)(t * 16 + ln) * D_DIM + k0 + laneHi * 16;
      acc[t] = wmma_bf16(A, loadB16(wr), acc[t]);
    }
  }
#pragma unroll
  for (int t = 0; t < 8; ++t)
#pragma unroll
    for (int r = 0; r < 8; ++r)
      mybufA[r + laneHi * 8][t * 16 + ln] = f2bf(acc[t][r]);

  // ---------------- Stage 2: q = (f @ Wq^T + bq) * 0.25 -------------------
  {
    v8f qa[8];
#pragma unroll
    for (int t = 0; t < 8; ++t)
#pragma unroll
      for (int r = 0; r < 8; ++r) qa[t][r] = bqv[t];
#pragma unroll
    for (int kc = 0; kc < 4; ++kc) {
      v16bf A = loadA_lds(mybufA[ln], kc * 32, laneHi);
#pragma unroll
      for (int t = 0; t < 8; ++t) {
        const __bf16* wr = Wqb + (size_t)(t * 16 + ln) * E_DIM + kc * 32 + laneHi * 16;
        qa[t] = wmma_bf16(A, loadB16(wr), qa[t]);
      }
    }
#pragma unroll
    for (int t = 0; t < 8; ++t)
#pragma unroll
      for (int r = 0; r < 8; ++r)
        mybufA[r + laneHi * 8][t * 16 + ln] = f2bf(qa[t][r] * 0.25f);
  }

  // ---------------- Attention (8 heads, 2 heads share a 32-K chunk) -------
  v8f oacc[8];
#pragma unroll
  for (int h = 0; h < 8; ++h)
#pragma unroll
    for (int r = 0; r < 8; ++r) oacc[h][r] = 0.f;

  for (int h = 0; h < 8; ++h) {
    v8f s[8];
#pragma unroll
    for (int t = 0; t < 8; ++t)
#pragma unroll
      for (int r = 0; r < 8; ++r) s[t][r] = 0.f;

    v16bf Aq = loadA_lds(mybufA[ln], (h >> 1) * 32, laneHi);
#pragma unroll
    for (int t = 0; t < 8; ++t) {
      const __bf16* kr = Karr + ((size_t)h * 128 + t * 16 + ln) * 32 + laneHi * 16;
      s[t] = wmma_bf16(Aq, loadB16(kr), s[t]);
    }

    // softmax over 128 labels (8 tiles x 16 lanes of this half-wave)
#pragma unroll
    for (int r = 0; r < 8; ++r) {
      float m = s[0][r];
#pragma unroll
      for (int t = 1; t < 8; ++t) m = fmaxf(m, s[t][r]);
      m = fmaxf(m, __shfl_xor(m, 1));
      m = fmaxf(m, __shfl_xor(m, 2));
      m = fmaxf(m, __shfl_xor(m, 4));
      m = fmaxf(m, __shfl_xor(m, 8));
      float e[8], sum = 0.f;
#pragma unroll
      for (int t = 0; t < 8; ++t) { e[t] = __expf(s[t][r] - m); sum += e[t]; }
      sum += __shfl_xor(sum, 1);
      sum += __shfl_xor(sum, 2);
      sum += __shfl_xor(sum, 4);
      sum += __shfl_xor(sum, 8);
      float inv = 1.0f / sum;
#pragma unroll
      for (int t = 0; t < 8; ++t)
        mybufB[r + laneHi * 8][t * 16 + ln] = f2bf(e[t] * inv);
    }

    // o[:, h*16:(h+1)*16] = attn @ vh   (K = 128 labels)
#pragma unroll
    for (int kc = 0; kc < 4; ++kc) {
      v16bf Aa = loadA_lds(mybufB[ln], kc * 32, laneHi);
      const __bf16* vr = Varr + ((size_t)h * 16 + ln) * 128 + kc * 32 + laneHi * 16;
      oacc[h] = wmma_bf16(Aa, loadB16(vr), oacc[h]);
    }
  }
#pragma unroll
  for (int h = 0; h < 8; ++h)
#pragma unroll
    for (int r = 0; r < 8; ++r)
      mybufA[r + laneHi * 8][h * 16 + ln] = f2bf(oacc[h][r]);

  // ---------------- Stage 5: o2 = o @ Wo^T + bo ---------------------------
  {
    v8f oa[8];
#pragma unroll
    for (int t = 0; t < 8; ++t)
#pragma unroll
      for (int r = 0; r < 8; ++r) oa[t][r] = bov[t];
#pragma unroll
    for (int kc = 0; kc < 4; ++kc) {
      v16bf A = loadA_lds(mybufA[ln], kc * 32, laneHi);
#pragma unroll
      for (int t = 0; t < 8; ++t) {
        const __bf16* wr = Wob + (size_t)(t * 16 + ln) * E_DIM + kc * 32 + laneHi * 16;
        oa[t] = wmma_bf16(A, loadB16(wr), oa[t]);
      }
    }
#pragma unroll
    for (int t = 0; t < 8; ++t)
#pragma unroll
      for (int r = 0; r < 8; ++r)
        mybufA[r + laneHi * 8][t * 16 + ln] = f2bf(oa[t][r]);
  }

  // ---------------- Stage 6: logits = o2 @ Wc^T + bc ----------------------
  {
    v8f lg[8];
#pragma unroll
    for (int t = 0; t < 8; ++t)
#pragma unroll
      for (int r = 0; r < 8; ++r) lg[t][r] = bcv[t];
#pragma unroll
    for (int kc = 0; kc < 4; ++kc) {
      v16bf A = loadA_lds(mybufA[ln], kc * 32, laneHi);
#pragma unroll
      for (int t = 0; t < 8; ++t) {
        const __bf16* wr = Wcb + (size_t)(t * 16 + ln) * E_DIM + kc * 32 + laneHi * 16;
        lg[t] = wmma_bf16(A, loadB16(wr), lg[t]);
      }
    }
#pragma unroll
    for (int t = 0; t < 8; ++t)
#pragma unroll
      for (int r = 0; r < 8; ++r)
        out[(size_t)(rowBase + r + laneHi * 8) * C_DIM + t * 16 + ln] = lg[t][r];
  }
}

// ---------------------------------------------------------------------------
// Workspace layout (bytes)
// ---------------------------------------------------------------------------
#define WS_WFB   0u
#define WS_WQB   1048576u
#define WS_WOB   1081344u
#define WS_WCB   1114112u
#define WS_KARR  1146880u
#define WS_VARR  1212416u

extern "C" void kernel_launch(void* const* d_in, const int* in_sizes, int n_in,
                              void* d_out, int out_size, void* d_ws, size_t ws_size,
                              hipStream_t stream) {
  const float* x         = (const float*)d_in[0];
  const float* label_emb = (const float*)d_in[1];
  const float* Wf        = (const float*)d_in[2];
  const float* bf        = (const float*)d_in[3];
  const float* Wq        = (const float*)d_in[4];
  const float* bq        = (const float*)d_in[5];
  const float* Wk        = (const float*)d_in[6];
  const float* bk        = (const float*)d_in[7];
  const float* Wv        = (const float*)d_in[8];
  const float* bv        = (const float*)d_in[9];
  const float* Wo        = (const float*)d_in[10];
  const float* bo        = (const float*)d_in[11];
  const float* Wc        = (const float*)d_in[12];
  const float* bc        = (const float*)d_in[13];
  float* out = (float*)d_out;
  char*  ws  = (char*)d_ws;

  __bf16* Wfb  = (__bf16*)(ws + WS_WFB);
  __bf16* Wqb  = (__bf16*)(ws + WS_WQB);
  __bf16* Wob  = (__bf16*)(ws + WS_WOB);
  __bf16* Wcb  = (__bf16*)(ws + WS_WCB);
  __bf16* Karr = (__bf16*)(ws + WS_KARR);
  __bf16* Varr = (__bf16*)(ws + WS_VARR);

  cvt_bf16_kernel<<<(E_DIM * D_DIM + 255) / 256, 256, 0, stream>>>(Wf, Wfb, E_DIM * D_DIM);
  cvt_bf16_kernel<<<64, 256, 0, stream>>>(Wq, Wqb, E_DIM * E_DIM);
  cvt_bf16_kernel<<<64, 256, 0, stream>>>(Wo, Wob, E_DIM * E_DIM);
  cvt_bf16_kernel<<<64, 256, 0, stream>>>(Wc, Wcb, E_DIM * E_DIM);
  prep_kv_kernel<<<64, 256, 0, stream>>>(label_emb, Wk, bk, Wv, bv, Karr, Varr);

  fused_kernel<<<B_ROWS / 64, 128, 0, stream>>>(x, bf, bq, bo, bc,
                                                Wfb, Wqb, Wob, Wcb,
                                                Karr, Varr, out);
}
